// MAB_20263655702989
// MI455X (gfx1250) — matrix operand order repaired and checked by reading, not compiled, not run on hardware
//
#include <hip/hip_runtime.h>
#include <stdint.h>

// ---------------------------------------------------------------------------
// Problem constants (match reference)
// ---------------------------------------------------------------------------
constexpr int kB  = 8;
constexpr int kS  = 2048;
constexpr int kD  = 256;     // DQ == DK == DV
constexpr int kH  = 4;
constexpr int kDH = 64;      // DV / H
constexpr int kBS = kB * kS; // 16384 rows
constexpr float kEps = 1e-5f;

typedef __bf16 bf16_t;
typedef __attribute__((ext_vector_type(16))) __bf16 v16bf;
typedef __attribute__((ext_vector_type(8)))  __bf16 v8bf;
typedef __attribute__((ext_vector_type(8)))  float  v8f;
typedef __attribute__((ext_vector_type(4)))  unsigned int v4u;
typedef __attribute__((ext_vector_type(4)))  int v4i;
typedef __attribute__((ext_vector_type(8)))  int v8i;

// ---------------------------------------------------------------------------
// WMMA helpers (CDNA5: V_WMMA_F32_16X16X32_BF16, wave32)
//
// A layout (16-bit, 16x32): lane holds row m = lane&15; half = lane>>4;
//   element i    -> K = half*16 + i     (16 contiguous K values; this code's order, not the ISA table order)
// B layout (32x16): lane holds col n = lane&15;
//   element i    -> K = (lane>>4)*16 + i     (16 contiguous K values; same order as A, not the ISA table order)
// C/D layout (16x16 f32, 8 VGPRs): elem r -> row M = r + 8*(lane>>4),
//   col N = lane&15.
// ---------------------------------------------------------------------------
__device__ __forceinline__ v8f wmma_bf16(v16bf a, v16bf b, v8f c) {
  return __builtin_amdgcn_wmma_f32_16x16x32_bf16(
      /*neg_a=*/false, a, /*neg_b=*/false, b,
      /*c_mod=*/(short)0, c, /*reuse_a=*/false, /*reuse_b=*/false);
}

__device__ __forceinline__ v16bf combine8(v8bf lo, v8bf hi) {
  v16bf r;
#pragma unroll
  for (int i = 0; i < 8; ++i) { r[i] = lo[i]; r[i + 8] = hi[i]; }
  return r;
}

// A fragment from a bf16 row pointer (row-major, K contiguous)
__device__ __forceinline__ v16bf load_a_bf16(const bf16_t* rowptr, int kb, int half) {
  v8bf lo = *(const v8bf*)(rowptr + kb + half * 16);
  v8bf hi = *(const v8bf*)(rowptr + kb + half * 16 + 8);
  return combine8(lo, hi);
}

// A fragment built from fp32 row data (convert to bf16 on the fly)
__device__ __forceinline__ v16bf load_a_f32(const float* rowptr, int kb, int half) {
  const float* p0 = rowptr + kb + half * 16;
  const float* p1 = rowptr + kb + half * 16 + 8;
  v16bf r;
#pragma unroll
  for (int i = 0; i < 8; ++i) {
    r[i]     = (bf16_t)p0[i];
    r[i + 8] = (bf16_t)p1[i];
  }
  return r;
}

// B fragment: 16 contiguous bf16 K-values for this lane's column.
__device__ __forceinline__ v16bf load_b16(const bf16_t* p) {
  v8bf lo = *(const v8bf*)(p);
  v8bf hi = *(const v8bf*)(p + 8);
  return combine8(lo, hi);
}

// ---------------------------------------------------------------------------
// Tensor Data Mover: 2-D bf16 tile (tile_d0 elems/row x tile_d1 rows),
// row stride `stride0` elements, global -> LDS at byte offset lds_off.
// D# per CDNA5 ISA ch.8: group0 = {count=1, lds_addr, global_addr, type=2},
// group1 = {mask=0, data_size=1(2B), tensor_dim == tile_dim (no OOB),
// dim0 stride}, groups 2/3 + extended group zero (<=2-D tile).
// Toolchain uses the 6-arg builtin: (v4u, v8i, v4i, v4i, v8i, i32 cpol).
// ---------------------------------------------------------------------------
__device__ __forceinline__ void tdm_load_2d_bf16(unsigned lds_off,
                                                 const bf16_t* gptr,
                                                 unsigned tile_d0,
                                                 unsigned tile_d1,
                                                 unsigned stride0) {
  unsigned long long ga = (unsigned long long)(uintptr_t)gptr;
  v4u g0;
  g0[0] = 1u;                                        // count=1 (user mode)
  g0[1] = lds_off;                                   // lds_addr (bytes)
  g0[2] = (unsigned)(ga & 0xffffffffu);              // global_addr[31:0]
  g0[3] = (unsigned)((ga >> 32) & 0x01ffffffu)       // global_addr[56:32]
        | (2u << 30);                                // type = 2 ("image")
  v8i g1;
  g1[0] = (int)(1u << 16);                           // data_size=1 -> 2 bytes
  g1[1] = (int)((tile_d0 & 0xffffu) << 16);          // tensor_dim0[15:0]
  g1[2] = (int)(((tile_d0 >> 16) & 0xffffu) | ((tile_d1 & 0xffffu) << 16));
  g1[3] = (int)(((tile_d1 >> 16) & 0xffffu) | ((tile_d0 & 0xffffu) << 16)); // tile_dim0
  g1[4] = (int)(tile_d1 & 0xffffu);                  // tile_dim1, tile_dim2=0
  g1[5] = (int)stride0;                              // tensor_dim0_stride[31:0]
  g1[6] = 0;
  g1[7] = 0;
  v4i z4 = {0, 0, 0, 0};
  v8i z8 = {0, 0, 0, 0, 0, 0, 0, 0};
  __builtin_amdgcn_tensor_load_to_lds(g0, g1, z4, z4, z8, 0);
}

__device__ __forceinline__ unsigned lds_offset_of(const void* p) {
  return (unsigned)(uintptr_t)p;   // generic LDS address: low 32 bits = LDS offset
}

// ---------------------------------------------------------------------------
// Kernel 0: transpose weights to bf16, n-major: Wt[n][k] = (bf16)W[k][n]
// ---------------------------------------------------------------------------
__global__ void prep_weights_kernel(const float* __restrict__ Wq,
                                    const float* __restrict__ Wk,
                                    const float* __restrict__ Wv,
                                    const float* __restrict__ Wo,
                                    bf16_t* __restrict__ Wqt,
                                    bf16_t* __restrict__ Wkt,
                                    bf16_t* __restrict__ Wvt,
                                    bf16_t* __restrict__ Wot) {
  int idx = blockIdx.x * 256 + threadIdx.x;   // 0..65535
  int n = idx >> 8;
  int k = idx & 255;
  size_t src = (size_t)k * kD + n;
  size_t dst = (size_t)n * kD + k;
  Wqt[dst] = (bf16_t)Wq[src];
  Wkt[dst] = (bf16_t)Wk[src];
  Wvt[dst] = (bf16_t)Wv[src];
  Wot[dst] = (bf16_t)Wo[src];
}

// ---------------------------------------------------------------------------
// Kernel 1: projections  q = Q@Wq+bq, k = K@Wk+bk, v = K@Wv+bv
// Block = 128 threads (4 waves). Each block: 64 rows x 64 cols.
// which (blockIdx.z): 0 -> q (fp32 + bf16, row-major)
//                     1 -> k (bf16 row-major)
//                     2 -> v (bf16 transposed [b][h][dh][s])
// ---------------------------------------------------------------------------
__global__ void proj_kernel(const float* __restrict__ Qin,
                            const float* __restrict__ Kin,
                            const bf16_t* __restrict__ Wqt,
                            const bf16_t* __restrict__ Wkt,
                            const bf16_t* __restrict__ Wvt,
                            const float* __restrict__ bq,
                            const float* __restrict__ bk,
                            const float* __restrict__ bv,
                            float* __restrict__ qf32,
                            bf16_t* __restrict__ qbf,
                            bf16_t* __restrict__ kbf,
                            bf16_t* __restrict__ vtbf) {
  const int which = blockIdx.z;
  const int ntile = blockIdx.y;              // cols [64*ntile, +64)
  const int mblk  = blockIdx.x;              // rows [64*mblk, +64)
  const int wave  = threadIdx.x >> 5;
  const int lane  = threadIdx.x & 31;
  const int half  = lane >> 4;
  const int l16   = lane & 15;

  const float*  src  = (which == 0) ? Qin : Kin;
  const bf16_t* Wt   = (which == 0) ? Wqt : ((which == 1) ? Wkt : Wvt);
  const float*  bias = (which == 0) ? bq  : ((which == 1) ? bk  : bv);

  const int arow_idx = mblk * 64 + wave * 16 + l16;
  const float* arow = src + (size_t)arow_idx * kD;

  v8f acc[4] = {v8f{}, v8f{}, v8f{}, v8f{}};

  for (int kb = 0; kb < kD; kb += 32) {
    v16bf a = load_a_f32(arow, kb, half);
#pragma unroll
    for (int t = 0; t < 4; ++t) {
      int n = ntile * 64 + t * 16 + l16;
      v16bf b = load_b16(Wt + (size_t)n * kD + kb + half * 16);
      acc[t] = wmma_bf16(a, b, acc[t]);
    }
  }

  const int mbase = mblk * 64 + wave * 16;
#pragma unroll
  for (int t = 0; t < 4; ++t) {
    int col = ntile * 64 + t * 16 + l16;
    float bcol = bias[col];
#pragma unroll
    for (int r = 0; r < 8; ++r) {
      int m = mbase + r + 8 * half;
      float v = acc[t][r] + bcol;
      if (which == 0) {
        qf32[(size_t)m * kD + col] = v;
        qbf [(size_t)m * kD + col] = (bf16_t)v;
      } else if (which == 1) {
        kbf [(size_t)m * kD + col] = (bf16_t)v;
      } else {
        int b  = m >> 11;          // m / S
        int s  = m & (kS - 1);
        int h  = col >> 6;         // col / DH
        int dh = col & (kDH - 1);
        vtbf[((size_t)(b * kH + h) * kDH + dh) * kS + s] = (bf16_t)v;
      }
    }
  }
}

// ---------------------------------------------------------------------------
// Kernel 2: flash attention with residual, TDM-staged K/V tiles.
// o[b,q,h,:] = q4[b,q,h,:] + softmax(q4·k4^T / 8) @ v4
// Block = 128 threads (4 waves) covers 64 queries of one (b,h); each wave
// owns a 16-query tile. Key loop in chunks of 32. K-chunk (32x64 bf16) and
// V-chunk (64x32 bf16, transposed layout) are DMA'd into double-buffered
// LDS by the Tensor Data Mover (issued by wave 0 only), so compute waves
// touch only LDS inside the loop.
// ---------------------------------------------------------------------------
__global__ void attn_kernel(const bf16_t* __restrict__ qbf,
                            const bf16_t* __restrict__ kbf,
                            const bf16_t* __restrict__ vtbf,
                            const float* __restrict__ qf32,
                            float* __restrict__ Obuf) {
  __shared__ __attribute__((aligned(16))) bf16_t lds_k[2][32][64]; // [buf][key][dh]
  __shared__ __attribute__((aligned(16))) bf16_t lds_v[2][64][32]; // [buf][dh][key]
  __shared__ __attribute__((aligned(16))) bf16_t lds_p[4][16][32];

  const int b    = blockIdx.z;
  const int h    = blockIdx.y;
  const int qt   = blockIdx.x;               // 0..31 (64-query tiles)
  const int wave = threadIdx.x >> 5;
  const int lane = threadIdx.x & 31;
  const int half = lane >> 4;
  const int l16  = lane & 15;
  const int q0   = qt * 64 + wave * 16;

  const bf16_t* qbase  = qbf + (size_t)b * kS * kD + h * kDH;
  const bf16_t* kgbase = kbf + (size_t)b * kS * kD + h * kDH;   // + kc*kD per chunk
  const bf16_t* vgbase = vtbf + (size_t)(b * kH + h) * kDH * kS; // + kc per chunk

  const unsigned k_lds0 = lds_offset_of(&lds_k[0][0][0]);
  const unsigned k_lds1 = lds_offset_of(&lds_k[1][0][0]);
  const unsigned v_lds0 = lds_offset_of(&lds_v[0][0][0]);
  const unsigned v_lds1 = lds_offset_of(&lds_v[1][0][0]);

  // Q fragments for this wave's 16 rows (dh windows 0..31 and 32..63)
  const bf16_t* qrow = qbase + (size_t)(q0 + l16) * kD;
  v16bf qa0 = load_a_bf16(qrow, 0, half);
  v16bf qa1 = load_a_bf16(qrow, 32, half);

  v8f o[4] = {v8f{}, v8f{}, v8f{}, v8f{}};
  float mrow[8], lrow[8];
#pragma unroll
  for (int r = 0; r < 8; ++r) { mrow[r] = -1e30f; lrow[r] = 0.0f; }

  const float scale = 0.125f;   // 1/sqrt(64)
  constexpr int nchunks = kS / 32;

  // Preload chunk 0 into buffer 0 (one TDM issue per tile, wave 0 only).
  if (wave == 0) {
    tdm_load_2d_bf16(k_lds0, kgbase, /*d0=*/64, /*d1=*/32, /*stride=*/kD);
    tdm_load_2d_bf16(v_lds0, vgbase, /*d0=*/32, /*d1=*/64, /*stride=*/kS);
  }

  for (int i = 0; i < nchunks; ++i) {
    const int buf = i & 1;
    if (wave == 0) {
      if (i + 1 < nchunks) {
        const int nk = (i + 1) * 32;
        tdm_load_2d_bf16(buf ? k_lds0 : k_lds1, kgbase + (size_t)nk * kD, 64, 32, kD);
        tdm_load_2d_bf16(buf ? v_lds0 : v_lds1, vgbase + nk, 32, 64, kS);
        __builtin_amdgcn_s_wait_tensorcnt(2);   // chunk i's 2 loads done (in-order)
      } else {
        __builtin_amdgcn_s_wait_tensorcnt(0);
      }
    }
    __syncthreads();   // chunk i visible in LDS to all waves

    // ---- scores: 16 queries x 32 keys, accumulate over dh=64 (LDS B frags) ----
    v8f s0 = v8f{}, s1 = v8f{};
    {
      const bf16_t* kr0 = &lds_k[buf][l16][0];
      v16bf b00 = load_b16(kr0 + half * 16);        // dh 0..31
      v16bf b01 = load_b16(kr0 + 32 + half * 16);   // dh 32..63
      s0 = wmma_bf16(qa0, b00, s0);
      s0 = wmma_bf16(qa1, b01, s0);
      const bf16_t* kr1 = &lds_k[buf][16 + l16][0];
      v16bf b10 = load_b16(kr1 + half * 16);
      v16bf b11 = load_b16(kr1 + 32 + half * 16);
      s1 = wmma_bf16(qa0, b10, s1);
      s1 = wmma_bf16(qa1, b11, s1);
    }

    // ---- online softmax (row = r + 8*half, cols across 16-lane group) ----
#pragma unroll
    for (int r = 0; r < 8; ++r) {
      float v0 = s0[r] * scale;
      float v1 = s1[r] * scale;
      float mx = fmaxf(v0, v1);
#pragma unroll
      for (int m = 1; m < 16; m <<= 1) mx = fmaxf(mx, __shfl_xor(mx, m, 32));
      float mnew  = fmaxf(mrow[r], mx);
      float alpha = __expf(mrow[r] - mnew);
      mrow[r] = mnew;
      float p0 = __expf(v0 - mnew);
      float p1 = __expf(v1 - mnew);
      float ps = p0 + p1;
#pragma unroll
      for (int m = 1; m < 16; m <<= 1) ps += __shfl_xor(ps, m, 32);
      lrow[r] = lrow[r] * alpha + ps;
#pragma unroll
      for (int t = 0; t < 4; ++t) o[t][r] *= alpha;
      lds_p[wave][r + 8 * half][l16]      = (bf16_t)p0;
      lds_p[wave][r + 8 * half][16 + l16] = (bf16_t)p1;
    }
    __syncthreads();

    // Re-read P in A-fragment layout (K = key index within chunk)
    v16bf pa = load_a_bf16(&lds_p[wave][l16][0], 0, half);
    __syncthreads();

    // ---- PV: 4 dh tiles of 16; V tile is [dh][key] so K is contiguous ----
#pragma unroll
    for (int t = 0; t < 4; ++t) {
      v16bf vb = load_b16(&lds_v[buf][t * 16 + l16][0] + half * 16);
      o[t] = wmma_bf16(pa, vb, o[t]);
    }
    __syncthreads();   // all waves done with buf before TDM overwrites it
  }

  // ---- epilogue: normalize, add residual q, store fp32 ----
#pragma unroll
  for (int t = 0; t < 4; ++t) {
#pragma unroll
    for (int r = 0; r < 8; ++r) {
      int m   = q0 + r + 8 * half;
      int col = h * kDH + t * 16 + l16;
      size_t idx = ((size_t)b * kS + m) * kD + col;
      Obuf[idx] = o[t][r] / lrow[r] + qf32[idx];
    }
  }
}

// ---------------------------------------------------------------------------
// Kernel 3: LayerNorm 0. One wave per 256-wide row.
// ---------------------------------------------------------------------------
__global__ void ln0_kernel(const float* __restrict__ Obuf,
                           const float* __restrict__ g0,
                           const float* __restrict__ beta0,
                           float* __restrict__ Xf,
                           bf16_t* __restrict__ Xbf) {
  int row  = blockIdx.x * (blockDim.x >> 5) + (threadIdx.x >> 5);
  int lane = threadIdx.x & 31;
  const float* rp = Obuf + (size_t)row * kD;
  float vals[8];
  float sum = 0.f, sq = 0.f;
#pragma unroll
  for (int j = 0; j < 8; ++j) {
    float x = rp[lane + 32 * j];
    vals[j] = x; sum += x; sq += x * x;
  }
#pragma unroll
  for (int m = 1; m < 32; m <<= 1) {
    sum += __shfl_xor(sum, m, 32);
    sq  += __shfl_xor(sq,  m, 32);
  }
  float mu  = sum * (1.0f / kD);
  float var = sq * (1.0f / kD) - mu * mu;
  float rs  = rsqrtf(var + kEps);
#pragma unroll
  for (int j = 0; j < 8; ++j) {
    int c = lane + 32 * j;
    float y = (vals[j] - mu) * rs * g0[c] + beta0[c];
    Xf [(size_t)row * kD + c] = y;
    Xbf[(size_t)row * kD + c] = (bf16_t)y;
  }
}

// ---------------------------------------------------------------------------
// Kernel 4: Y = X + relu(X@Wo + bo); out = LN1(Y)
// Block = 128 threads (4 waves); block owns 16 rows x 256 cols.
// Wave w computes cols [64w, 64w+64); LN1 fused via a 16x256 LDS tile.
// ---------------------------------------------------------------------------
__global__ void out_kernel(const bf16_t* __restrict__ Xbf,
                           const float* __restrict__ Xf,
                           const bf16_t* __restrict__ Wot,
                           const float* __restrict__ bo,
                           const float* __restrict__ g1,
                           const float* __restrict__ beta1,
                           float* __restrict__ Out) {
  __shared__ __attribute__((aligned(16))) float ytile[16][kD];

  const int mblk = blockIdx.x;               // rows [16*mblk, +16)
  const int wave = threadIdx.x >> 5;
  const int lane = threadIdx.x & 31;
  const int half = lane >> 4;
  const int l16  = lane & 15;

  const bf16_t* arow = Xbf + (size_t)(mblk * 16 + l16) * kD;

  v8f acc[4] = {v8f{}, v8f{}, v8f{}, v8f{}};
  for (int kb = 0; kb < kD; kb += 32) {
    v16bf a = load_a_bf16(arow, kb, half);
#pragma unroll
    for (int t = 0; t < 4; ++t) {
      int n = wave * 64 + t * 16 + l16;
      v16bf b = load_b16(Wot + (size_t)n * kD + kb + half * 16);
      acc[t] = wmma_bf16(a, b, acc[t]);
    }
  }

#pragma unroll
  for (int t = 0; t < 4; ++t) {
    int col = wave * 64 + t * 16 + l16;
    float bcol = bo[col];
#pragma unroll
    for (int r = 0; r < 8; ++r) {
      int m = r + 8 * half;                  // local row 0..15
      float x = Xf[(size_t)(mblk * 16 + m) * kD + col];
      float y = acc[t][r] + bcol;
      ytile[m][col] = x + fmaxf(y, 0.0f);
    }
  }
  __syncthreads();

  // LN1: wave w handles local rows [4w, 4w+4)
#pragma unroll
  for (int rr = 0; rr < 4; ++rr) {
    int m = wave * 4 + rr;
    float vals[8];
    float sum = 0.f, sq = 0.f;
#pragma unroll
    for (int j = 0; j < 8; ++j) {
      float x = ytile[m][lane + 32 * j];
      vals[j] = x; sum += x; sq += x * x;
    }
#pragma unroll
    for (int mm = 1; mm < 32; mm <<= 1) {
      sum += __shfl_xor(sum, mm, 32);
      sq  += __shfl_xor(sq,  mm, 32);
    }
    float mu  = sum * (1.0f / kD);
    float var = sq * (1.0f / kD) - mu * mu;
    float rs  = rsqrtf(var + kEps);
#pragma unroll
    for (int j = 0; j < 8; ++j) {
      int c = lane + 32 * j;
      Out[(size_t)(mblk * 16 + m) * kD + c] = (vals[j] - mu) * rs * g1[c] + beta1[c];
    }
  }
}

// ---------------------------------------------------------------------------
// Workspace layout (bytes)
// ---------------------------------------------------------------------------
constexpr size_t OFF_QF32 = 0;                                   // 16 MB fp32 q
constexpr size_t OFF_OBUF = OFF_QF32 + (size_t)kBS * kD * 4;     // 16 MB fp32 attn out
constexpr size_t OFF_XF   = OFF_OBUF + (size_t)kBS * kD * 4;     // 16 MB fp32 LN0 out
constexpr size_t OFF_QBF  = OFF_XF   + (size_t)kBS * kD * 4;     //  8 MB bf16 q
constexpr size_t OFF_KBF  = OFF_QBF  + (size_t)kBS * kD * 2;     //  8 MB bf16 k
constexpr size_t OFF_VT   = OFF_KBF  + (size_t)kBS * kD * 2;     //  8 MB bf16 v (transposed)
constexpr size_t OFF_XBF  = OFF_VT   + (size_t)kBS * kD * 2;     //  8 MB bf16 LN0 out
constexpr size_t OFF_WQT  = OFF_XBF  + (size_t)kBS * kD * 2;
constexpr size_t OFF_WKT  = OFF_WQT  + (size_t)kD * kD * 2;
constexpr size_t OFF_WVT  = OFF_WKT  + (size_t)kD * kD * 2;
constexpr size_t OFF_WOT  = OFF_WVT  + (size_t)kD * kD * 2;

extern "C" void kernel_launch(void* const* d_in, const int* in_sizes, int n_in,
                              void* d_out, int out_size, void* d_ws, size_t ws_size,
                              hipStream_t stream) {
  const float* Q     = (const float*)d_in[0];
  const float* K     = (const float*)d_in[1];
  const float* Wq    = (const float*)d_in[2];
  const float* bq    = (const float*)d_in[3];
  const float* Wk    = (const float*)d_in[4];
  const float* bk    = (const float*)d_in[5];
  const float* Wv    = (const float*)d_in[6];
  const float* bv    = (const float*)d_in[7];
  const float* Wo    = (const float*)d_in[8];
  const float* bo    = (const float*)d_in[9];
  const float* g0    = (const float*)d_in[10];
  const float* beta0 = (const float*)d_in[11];
  const float* g1    = (const float*)d_in[12];
  const float* beta1 = (const float*)d_in[13];
  float* Out = (float*)d_out;

  char* ws = (char*)d_ws;
  float*  qf32 = (float*)(ws + OFF_QF32);
  float*  Obuf = (float*)(ws + OFF_OBUF);
  float*  Xf   = (float*)(ws + OFF_XF);
  bf16_t* qbf  = (bf16_t*)(ws + OFF_QBF);
  bf16_t* kbf  = (bf16_t*)(ws + OFF_KBF);
  bf16_t* vt   = (bf16_t*)(ws + OFF_VT);
  bf16_t* Xbf  = (bf16_t*)(ws + OFF_XBF);
  bf16_t* Wqt  = (bf16_t*)(ws + OFF_WQT);
  bf16_t* Wkt  = (bf16_t*)(ws + OFF_WKT);
  bf16_t* Wvt  = (bf16_t*)(ws + OFF_WVT);
  bf16_t* Wot  = (bf16_t*)(ws + OFF_WOT);

  // 0) weight transpose + bf16 cast
  prep_weights_kernel<<<dim3(256), dim3(256), 0, stream>>>(
      Wq, Wk, Wv, Wo, Wqt, Wkt, Wvt, Wot);

  // 1) projections q,k,v (grid.z selects which)
  proj_kernel<<<dim3(kBS / 64, kD / 64, 3), dim3(128), 0, stream>>>(
      Q, K, Wqt, Wkt, Wvt, bq, bk, bv, qf32, qbf, kbf, vt);

  // 2) flash attention + residual (TDM double-buffered K/V staging)
  attn_kernel<<<dim3(kS / 64, kH, kB), dim3(128), 0, stream>>>(
      qbf, kbf, vt, qf32, Obuf);

  // 3) LayerNorm 0
  ln0_kernel<<<dim3(kBS / 8), dim3(256), 0, stream>>>(
      Obuf, g0, beta0, Xf, Xbf);

  // 4) GEMM + bias + ReLU + residual + LayerNorm 1
  out_kernel<<<dim3(kBS / 16), dim3(128), 0, stream>>>(
      Xbf, Xf, Wot, bo, g1, beta1, Out);
}
